// TemporalMoEViT_Encoder_23553600651636
// MI455X (gfx1250) — matrix operand order repaired and checked by reading, not compiled, not run on hardware
//
#include <hip/hip_runtime.h>
#include <math.h>

// ---------------------------------------------------------------------------
// Types / helpers
// ---------------------------------------------------------------------------
typedef __bf16 bf16;
typedef __attribute__((ext_vector_type(16))) __bf16 v16bf;
typedef __attribute__((ext_vector_type(8)))  float  v8f;

union F16x16 { v16bf v; uint4 q[2]; __bf16 h[16]; };

#ifndef __has_builtin
#define __has_builtin(x) 0
#endif
#if __has_builtin(__builtin_amdgcn_global_load_async_to_lds_b128)
#define HAVE_ASYNC_LDS 1
#else
#define HAVE_ASYNC_LDS 0
#endif

#if HAVE_ASYNC_LDS
typedef int v4i __attribute__((vector_size(16)));
typedef __attribute__((address_space(1))) v4i* as1v4;
typedef __attribute__((address_space(3))) v4i* as3v4;
__device__ __forceinline__ void async_cp16(const bf16* g, bf16* l) {
  // 16B per lane, global -> LDS, tracked by ASYNCcnt
  __builtin_amdgcn_global_load_async_to_lds_b128(
      (as1v4)(unsigned long long)(uintptr_t)g,
      (as3v4)(unsigned)(uintptr_t)l, 0, 0);
}
__device__ __forceinline__ void wait_async0() {
#if __has_builtin(__builtin_amdgcn_s_wait_asynccnt)
  __builtin_amdgcn_s_wait_asynccnt(0);
#else
  asm volatile("s_wait_asynccnt 0x0" ::: "memory");
#endif
}
#endif

__device__ __forceinline__ bf16 f2bf(float f) {
  unsigned u = __builtin_bit_cast(unsigned, f);
  unsigned r = (u + 0x7fffu + ((u >> 16) & 1u)) >> 16;   // RNE
  return __builtin_bit_cast(bf16, (unsigned short)r);
}
__device__ __forceinline__ float bf2f(bf16 b) {
  unsigned short s = __builtin_bit_cast(unsigned short, b);
  unsigned u = ((unsigned)s) << 16;
  return __builtin_bit_cast(float, u);
}
__device__ __forceinline__ float redmax16(float v) {
  v = fmaxf(v, __shfl_xor(v, 1, 32));
  v = fmaxf(v, __shfl_xor(v, 2, 32));
  v = fmaxf(v, __shfl_xor(v, 4, 32));
  v = fmaxf(v, __shfl_xor(v, 8, 32));
  return v;
}
__device__ __forceinline__ float redsum16(float v) {
  v += __shfl_xor(v, 1, 32);
  v += __shfl_xor(v, 2, 32);
  v += __shfl_xor(v, 4, 32);
  v += __shfl_xor(v, 8, 32);
  return v;
}

// Model constants
#define S_TOK 1568
#define DMODEL 768
#define HMAXW 3840
#define NHEAD 12

// ---------------------------------------------------------------------------
// Small utility kernels
// ---------------------------------------------------------------------------
__global__ void k_cvt_bf16(const float* __restrict__ src, bf16* __restrict__ dst, int n) {
  int i = blockIdx.x * blockDim.x + threadIdx.x;
  if (i < n) dst[i] = f2bf(src[i]);
}

__global__ void k_zero_i32(int* __restrict__ p, int n) {
  int i = blockIdx.x * blockDim.x + threadIdx.x;
  if (i < n) p[i] = 0;
}

// Patchify: video [1,8,3,224,224] -> P[s=1568][c*256+py*16+px] bf16
__global__ void k_patchify(const float* __restrict__ vf, bf16* __restrict__ P) {
  int idx = blockIdx.x * blockDim.x + threadIdx.x;
  if (idx >= S_TOK * DMODEL) return;
  int s = idx / DMODEL, cidx = idx - s * DMODEL;
  int t  = s / 196;
  int pr = (s % 196) / 14;
  int pc = s % 14;
  int c  = cidx >> 8;
  int py = (cidx >> 4) & 15;
  int px = cidx & 15;
  size_t off = (((size_t)(t * 3 + c) * 224 + pr * 16 + py) * 224) + pc * 16 + px;
  P[idx] = f2bf(vf[off]);
}

// LayerNorm over D=768, one row per block of 256 threads.
__global__ __launch_bounds__(256) void k_ln(const float* __restrict__ x,
                                            const float* __restrict__ g,
                                            const float* __restrict__ b,
                                            bf16* __restrict__ ob,
                                            float* __restrict__ of) {
  int row = blockIdx.x, tid = threadIdx.x;
  const float* xr = x + (size_t)row * DMODEL;
  float v0 = xr[tid], v1 = xr[tid + 256], v2 = xr[tid + 512];
  float s  = v0 + v1 + v2;
  float sq = v0 * v0 + v1 * v1 + v2 * v2;
  for (int m = 1; m < 32; m <<= 1) {
    s  += __shfl_xor(s,  m, 32);
    sq += __shfl_xor(sq, m, 32);
  }
  __shared__ float rs[8], rq[8];
  int lane = tid & 31, wave = tid >> 5;
  if (lane == 0) { rs[wave] = s; rq[wave] = sq; }
  __syncthreads();
  float ts = 0.f, tq = 0.f;
#pragma unroll
  for (int w = 0; w < 8; w++) { ts += rs[w]; tq += rq[w]; }
  float mean = ts * (1.f / DMODEL);
  float var  = tq * (1.f / DMODEL) - mean * mean;
  float rstd = rsqrtf(var + 1e-5f);
#pragma unroll
  for (int j = 0; j < 3; j++) {
    int col = tid + j * 256;
    float y = (xr[col] - mean) * rstd * g[col] + b[col];
    if (ob) ob[(size_t)row * DMODEL + col] = f2bf(y);
    if (of) of[(size_t)row * DMODEL + col] = y;
  }
}

// Router: one wave per token. Softmax over 4 logits, top-2, renormalized
// gates, append token to per-expert lists via atomics.
__global__ __launch_bounds__(32) void k_router(const bf16* __restrict__ h,
                                               const float* __restrict__ rw,
                                               int* __restrict__ counts,
                                               int* __restrict__ lists,
                                               float* __restrict__ glists) {
  int tok = blockIdx.x, lane = threadIdx.x;
  const bf16* hr = h + (size_t)tok * DMODEL;
  float a0 = 0.f, a1 = 0.f, a2 = 0.f, a3 = 0.f;
  for (int d = lane; d < DMODEL; d += 32) {
    float hv = bf2f(hr[d]);
    const float* r = rw + d * 4;
    a0 += hv * r[0]; a1 += hv * r[1]; a2 += hv * r[2]; a3 += hv * r[3];
  }
  for (int m = 1; m < 32; m <<= 1) {
    a0 += __shfl_xor(a0, m, 32);
    a1 += __shfl_xor(a1, m, 32);
    a2 += __shfl_xor(a2, m, 32);
    a3 += __shfl_xor(a3, m, 32);
  }
  if (lane == 0) {
    float lg[4] = {a0, a1, a2, a3};
    float mx = fmaxf(fmaxf(lg[0], lg[1]), fmaxf(lg[2], lg[3]));
    float pe[4]; float den = 0.f;
#pragma unroll
    for (int e = 0; e < 4; e++) { pe[e] = expf(lg[e] - mx); den += pe[e]; }
#pragma unroll
    for (int e = 0; e < 4; e++) pe[e] /= den;
    int i0 = 0;
#pragma unroll
    for (int e = 1; e < 4; e++) if (pe[e] > pe[i0]) i0 = e;
    int i1 = -1;
#pragma unroll
    for (int e = 0; e < 4; e++) if (e != i0 && (i1 < 0 || pe[e] > pe[i1])) i1 = e;
    float gs = pe[i0] + pe[i1];
    int p0 = atomicAdd(&counts[i0], 1);
    lists[i0 * S_TOK + p0]  = tok;
    glists[i0 * S_TOK + p0] = pe[i0] / gs;
    int p1 = atomicAdd(&counts[i1], 1);
    lists[i1 * S_TOK + p1]  = tok;
    glists[i1 * S_TOK + p1] = pe[i1] / gs;
  }
}

// ---------------------------------------------------------------------------
// Generic bf16 WMMA GEMM: C = act(A*B + bias) * rowScale + add0
//   A [M,K] bf16 row-major (optional row gather), B [K,N] bf16 row-major.
//   Tile 128x128x32, 8 waves (4 M x 2 N); each wave owns 32x64 = 2x4 WMMA
//   fragments -> 8 v_wmma per K-step from 6 LDS fragment loads.
// ---------------------------------------------------------------------------
__global__ __launch_bounds__(256) void k_gemm(
    const bf16* __restrict__ A, const bf16* __restrict__ B,
    const float* __restrict__ bias, const float* __restrict__ add0,
    float* __restrict__ Cf, bf16* __restrict__ Cb,
    int M, const int* __restrict__ Mdev,
    int N, int K, int lda, int ldb, int ldc,
    const int* __restrict__ rowIdxA, const int* __restrict__ rowIdxC,
    const float* __restrict__ rowScale, int doGelu) {
  constexpr int LP = 40;                    // LDS pitch (halves), 16B aligned
  __shared__ bf16 As[128 * LP];             // [m][k]
  __shared__ bf16 Bs[128 * LP];             // transposed: [n][k]
  int Mc = Mdev ? Mdev[0] : M;
  int m0 = blockIdx.x * 128;
  if (m0 >= Mc) return;
  int n0 = blockIdx.y * 128;
  int tid  = threadIdx.x;
  int lane = tid & 31, wave = tid >> 5;
  int mw = wave >> 1, nw = wave & 1;        // 4 M-waves x 2 N-waves
  int mr = lane & 15, hi = lane >> 4;

  v8f acc[2][4];
#pragma unroll
  for (int mi = 0; mi < 2; mi++)
#pragma unroll
    for (int nt = 0; nt < 4; nt++)
#pragma unroll
      for (int r = 0; r < 8; r++) acc[mi][nt][r] = 0.f;

  // A staging: thread -> (row tid>>1, 16 halves at (tid&1)*16)
  int ar = tid >> 1, ak = (tid & 1) * 16;
  long aoff = -1;
  {
    int arow = m0 + ar;
    if (arow < Mc) aoff = (long)(rowIdxA ? rowIdxA[arow] : arow) * lda;
  }
  // B staging: thread -> (k row tid>>3, 16 halves at (tid&7)*16), transposed
  int bk = tid >> 3, bn = (tid & 7) * 16;

  if (aoff < 0) {                 // out-of-range rows: zero once, reused all K
    uint4 z = make_uint4(0, 0, 0, 0);
    *(uint4*)&As[ar * LP + ak]     = z;
    *(uint4*)&As[ar * LP + ak + 8] = z;
  }

  for (int k0 = 0; k0 < K; k0 += 32) {
    if (aoff >= 0) {
#if HAVE_ASYNC_LDS
      async_cp16(A + aoff + k0 + ak,     &As[ar * LP + ak]);
      async_cp16(A + aoff + k0 + ak + 8, &As[ar * LP + ak + 8]);
#else
      *(uint4*)&As[ar * LP + ak]     = *(const uint4*)(A + aoff + k0 + ak);
      *(uint4*)&As[ar * LP + ak + 8] = *(const uint4*)(A + aoff + k0 + ak + 8);
#endif
    }
    uint4 bv0 = *(const uint4*)(B + (long)(k0 + bk) * ldb + n0 + bn);
    uint4 bv1 = *(const uint4*)(B + (long)(k0 + bk) * ldb + n0 + bn + 8);
    const bf16* bh0 = (const bf16*)&bv0;
    const bf16* bh1 = (const bf16*)&bv1;
#pragma unroll
    for (int j = 0; j < 8; j++) {
      Bs[(bn + j) * LP + bk]     = bh0[j];
      Bs[(bn + 8 + j) * LP + bk] = bh1[j];
    }
    if (k0 + 32 < K) {
      if (aoff >= 0) __builtin_prefetch(A + aoff + k0 + 32 + ak, 0, 1);
      __builtin_prefetch(B + (long)(k0 + 32 + bk) * ldb + n0 + bn, 0, 1);
    }
#if HAVE_ASYNC_LDS
    wait_async0();
#endif
    __syncthreads();

    F16x16 af[2];
#pragma unroll
    for (int mi = 0; mi < 2; mi++) {
      const bf16* ap = &As[(mw * 32 + mi * 16 + mr) * LP];
      af[mi].q[0] = *(const uint4*)(ap + hi * 8);
      af[mi].q[1] = *(const uint4*)(ap + 16 + hi * 8);
    }
#pragma unroll
    for (int nt = 0; nt < 4; nt++) {
      F16x16 bfrag;
      const bf16* bp = &Bs[(nw * 64 + nt * 16 + mr) * LP];
      bfrag.q[0] = *(const uint4*)(bp + hi * 8);
      bfrag.q[1] = *(const uint4*)(bp + 16 + hi * 8);
#pragma unroll
      for (int mi = 0; mi < 2; mi++) {
        acc[mi][nt] = __builtin_amdgcn_wmma_f32_16x16x32_bf16(
            false, af[mi].v, false, bfrag.v, (short)0, acc[mi][nt], false, false);
      }
    }
    __syncthreads();
  }

#pragma unroll
  for (int mi = 0; mi < 2; mi++) {
#pragma unroll
    for (int nt = 0; nt < 4; nt++) {
      int col = n0 + nw * 64 + nt * 16 + mr;
#pragma unroll
      for (int r = 0; r < 8; r++) {
        int crow = m0 + mw * 32 + mi * 16 + hi * 8 + r;
        if (crow < Mc) {
          float v = acc[mi][nt][r];
          if (bias) v += bias[col];
          if (doGelu) v = 0.5f * v * (1.f + erff(v * 0.70710678118654752f));
          if (rowScale) v *= rowScale[crow];
          long orow = rowIdxC ? (long)rowIdxC[crow] : (long)crow;
          if (add0) v += add0[orow * ldc + col];
          if (Cf) Cf[orow * ldc + col] = v;
          if (Cb) Cb[(long)crow * ldc + col] = f2bf(v);
        }
      }
    }
  }
}

// ---------------------------------------------------------------------------
// Flash attention: block = 2 waves (64 thr), each wave owns 16 query rows.
// Streams keys/values in chunks of 32 tokens. qkv bf16 [S, 2304] packed Q|K|V.
// ---------------------------------------------------------------------------
__global__ __launch_bounds__(64) void k_attn(const bf16* __restrict__ qkv,
                                             bf16* __restrict__ out, int S) {
  __shared__ bf16 Vs[32 * 64];
  __shared__ bf16 Ps[2][16 * 32];
  int wave = threadIdx.x >> 5, lane = threadIdx.x & 31;
  int mr = lane & 15, hi = lane >> 4;
  int q0 = blockIdx.x * 32 + wave * 16;
  int head = blockIdx.y;
  int qb = head * 64;

  // Q fragments (A-layout), loaded once, direct from global.
  F16x16 qa[2];
  {
    const bf16* qp = qkv + (size_t)(q0 + mr) * 2304 + qb;
#pragma unroll
    for (int c = 0; c < 2; c++) {
      qa[c].q[0] = *(const uint4*)(qp + c * 32 + hi * 8);
      qa[c].q[1] = *(const uint4*)(qp + c * 32 + 16 + hi * 8);
    }
  }
  v8f o[4];
#pragma unroll
  for (int t = 0; t < 4; t++)
#pragma unroll
    for (int r = 0; r < 8; r++) o[t][r] = 0.f;
  float rm[8], rl[8];
#pragma unroll
  for (int r = 0; r < 8; r++) { rm[r] = -1e30f; rl[r] = 0.f; }

  for (int kb = 0; kb < S; kb += 32) {
    // Stage V chunk [32 tok][64 d] into LDS (layout-preserving 16B copies).
    {
      int tok = threadIdx.x >> 1, part = threadIdx.x & 1;
      const bf16* sp = qkv + (size_t)(kb + tok) * 2304 + 1536 + qb + part * 32;
      bf16* dp = &Vs[tok * 64 + part * 32];
#if HAVE_ASYNC_LDS
#pragma unroll
      for (int i = 0; i < 4; i++) async_cp16(sp + i * 8, dp + i * 8);
      wait_async0();
#else
#pragma unroll
      for (int i = 0; i < 4; i++)
        ((uint4*)dp)[i] = ((const uint4*)sp)[i];
#endif
    }
    __syncthreads();

    // Scores: S[16q x 32k] = Q(16x64) * K^T(64x32), K frags direct from global.
    v8f s0, s1;
#pragma unroll
    for (int r = 0; r < 8; r++) { s0[r] = 0.f; s1[r] = 0.f; }
#pragma unroll
    for (int c = 0; c < 2; c++) {
      F16x16 kf0, kf1;
      const bf16* kp0 = qkv + (size_t)(kb + mr) * 2304 + 768 + qb + c * 32;
      kf0.q[0] = *(const uint4*)(kp0 + hi * 8);
      kf0.q[1] = *(const uint4*)(kp0 + 16 + hi * 8);
      const bf16* kp1 =
          qkv + (size_t)(kb + 16 + mr) * 2304 + 768 + qb + c * 32;
      kf1.q[0] = *(const uint4*)(kp1 + hi * 8);
      kf1.q[1] = *(const uint4*)(kp1 + 16 + hi * 8);
      s0 = __builtin_amdgcn_wmma_f32_16x16x32_bf16(false, qa[c].v, false, kf0.v,
                                                   (short)0, s0, false, false);
      s1 = __builtin_amdgcn_wmma_f32_16x16x32_bf16(false, qa[c].v, false, kf1.v,
                                                   (short)0, s1, false, false);
    }

    // Streaming softmax (rows live in one 16-lane half; xor<16 stays in-half).
    float p0[8], p1[8], nf[8];
#pragma unroll
    for (int r = 0; r < 8; r++) {
      float x0 = s0[r] * 0.125f, x1 = s1[r] * 0.125f;   // 1/sqrt(64)
      float mx = redmax16(fmaxf(x0, x1));
      float nm = fmaxf(rm[r], mx);
      float fac = expf(rm[r] - nm);
      float e0 = expf(x0 - nm), e1 = expf(x1 - nm);
      float sm = redsum16(e0 + e1);
      rl[r] = rl[r] * fac + sm;
      rm[r] = nm;
      p0[r] = e0; p1[r] = e1; nf[r] = fac;
    }
#pragma unroll
    for (int t = 0; t < 4; t++)
#pragma unroll
      for (int r = 0; r < 8; r++) o[t][r] *= nf[r];

    // P: C-layout -> A-layout via LDS.
#pragma unroll
    for (int r = 0; r < 8; r++) {
      Ps[wave][(hi * 8 + r) * 32 + mr]      = f2bf(p0[r]);
      Ps[wave][(hi * 8 + r) * 32 + 16 + mr] = f2bf(p1[r]);
    }
    __syncthreads();
    F16x16 pa;
    {
      const bf16* pp = &Ps[wave][mr * 32];
      pa.q[0] = *(const uint4*)(pp + hi * 8);
      pa.q[1] = *(const uint4*)(pp + 16 + hi * 8);
    }
    // O += P(16x32) * V(32x64)
#pragma unroll
    for (int nt = 0; nt < 4; nt++) {
      F16x16 vb;
#pragma unroll
      for (int j = 0; j < 16; j++) vb.h[j] = Vs[((j & 8) * 2 + hi * 8 + (j & 7)) * 64 + nt * 16 + mr];
      o[nt] = __builtin_amdgcn_wmma_f32_16x16x32_bf16(false, pa.v, false, vb.v,
                                                      (short)0, o[nt], false, false);
    }
    __syncthreads();
  }

#pragma unroll
  for (int nt = 0; nt < 4; nt++)
#pragma unroll
    for (int r = 0; r < 8; r++) {
      int tokr = q0 + hi * 8 + r;
      out[(size_t)tokr * 768 + qb + nt * 16 + mr] = f2bf(o[nt][r] / rl[r]);
    }
}

// ---------------------------------------------------------------------------
// Host orchestration
// ---------------------------------------------------------------------------
extern "C" void kernel_launch(void* const* d_in, const int* in_sizes, int n_in,
                              void* d_out, int out_size, void* d_ws, size_t ws_size,
                              hipStream_t stream) {
  (void)in_sizes; (void)n_in; (void)out_size; (void)ws_size;
  const float* video    = (const float*)d_in[0];
  const float* patch_w  = (const float*)d_in[1];
  const float* patch_b  = (const float*)d_in[2];
  const float* pos_emb  = (const float*)d_in[3];
  const float* ln1_g    = (const float*)d_in[4];
  const float* ln1_b    = (const float*)d_in[5];
  const float* qkv_w    = (const float*)d_in[6];
  const float* qkv_b    = (const float*)d_in[7];
  const float* proj_w   = (const float*)d_in[8];
  const float* proj_b   = (const float*)d_in[9];
  const float* ln2_g    = (const float*)d_in[10];
  const float* ln2_b    = (const float*)d_in[11];
  const float* router_w = (const float*)d_in[12];
  const float* w1       = (const float*)d_in[13];
  const float* b1       = (const float*)d_in[14];
  const float* w2       = (const float*)d_in[15];
  const float* w2b      = (const float*)d_in[16];
  const float* lnf_g    = (const float*)d_in[17];
  const float* lnf_b    = (const float*)d_in[18];

  const int S = S_TOK, D = DMODEL;
  static const int HIDv[4] = {1536, 2304, 3072, 3840};

  char* p = (char*)d_ws;
  auto alloc = [&](size_t bytes) -> void* {
    void* r = (void*)p;
    p += (bytes + 255) & ~(size_t)255;
    return r;
  };
  float* x_f32   = (float*)alloc((size_t)S * D * 4);
  bf16*  h_bf    = (bf16*)alloc((size_t)S * D * 2);
  bf16*  qkv_bf  = (bf16*)alloc((size_t)S * 3 * D * 2);
  bf16*  attn_bf = (bf16*)alloc((size_t)S * D * 2);
  bf16*  hid_bf  = (bf16*)alloc((size_t)S * HMAXW * 2);
  bf16*  pm_bf   = (bf16*)alloc((size_t)S * D * 2);
  bf16*  wb_pat  = (bf16*)alloc((size_t)D * D * 2);
  bf16*  wb_qkv  = (bf16*)alloc((size_t)D * 3 * D * 2);
  bf16*  wb_proj = (bf16*)alloc((size_t)D * D * 2);
  bf16*  wb_w1   = (bf16*)alloc((size_t)D * HMAXW * 2);
  bf16*  wb_w2   = (bf16*)alloc((size_t)HMAXW * D * 2);
  int*   counts  = (int*)alloc(4 * sizeof(int));
  int*   lists   = (int*)alloc((size_t)4 * S * sizeof(int));
  float* glists  = (float*)alloc((size_t)4 * S * sizeof(float));

  auto cvt = [&](const float* src, bf16* dst, int n) {
    k_cvt_bf16<<<(n + 255) / 256, 256, 0, stream>>>(src, dst, n);
  };
  auto gemm = [&](const bf16* A, const bf16* B, const float* bias,
                  const float* add0, float* Cf, bf16* Cb, int M,
                  const int* Mdev, int N, int K, int lda, int ldb, int ldc,
                  const int* rIA, const int* rIC, const float* rS, int gelu) {
    dim3 g((M + 127) / 128, N / 128);
    k_gemm<<<g, 256, 0, stream>>>(A, B, bias, add0, Cf, Cb, M, Mdev, N, K, lda,
                                  ldb, ldc, rIA, rIC, rS, gelu);
  };

  // ---- Patch embedding: x = patches @ W + b + pos ----
  cvt(patch_w, wb_pat, D * D);
  k_patchify<<<(S * D + 255) / 256, 256, 0, stream>>>(video, pm_bf);
  gemm(pm_bf, wb_pat, patch_b, pos_emb, x_f32, nullptr, S, nullptr, D, D, D, D,
       D, nullptr, nullptr, nullptr, 0);

  for (int l = 0; l < 2; l++) {
    cvt(qkv_w + (size_t)l * D * 3 * D, wb_qkv, D * 3 * D);
    cvt(proj_w + (size_t)l * D * D, wb_proj, D * D);

    // Attention block
    k_ln<<<S, 256, 0, stream>>>(x_f32, ln1_g + l * D, ln1_b + l * D, h_bf,
                                nullptr);
    gemm(h_bf, wb_qkv, qkv_b + l * 3 * D, nullptr, nullptr, qkv_bf, S, nullptr,
         3 * D, D, D, 3 * D, 3 * D, nullptr, nullptr, nullptr, 0);
    k_attn<<<dim3(S / 32, NHEAD), 64, 0, stream>>>(qkv_bf, attn_bf, S);
    gemm(attn_bf, wb_proj, proj_b + l * D, x_f32, x_f32, nullptr, S, nullptr, D,
         D, D, D, D, nullptr, nullptr, nullptr, 0);

    // MoE block (top-2 of 4 heterogeneous experts)
    k_ln<<<S, 256, 0, stream>>>(x_f32, ln2_g + l * D, ln2_b + l * D, h_bf,
                                nullptr);
    k_zero_i32<<<1, 32, 0, stream>>>(counts, 4);
    k_router<<<S, 32, 0, stream>>>(h_bf, router_w + (size_t)l * D * 4, counts,
                                   lists, glists);
    for (int e = 0; e < 4; e++) {
      int He = HIDv[e];
      cvt(w1 + ((size_t)(l * 4 + e)) * D * HMAXW, wb_w1, D * HMAXW);
      cvt(w2 + ((size_t)(l * 4 + e)) * HMAXW * D, wb_w2, HMAXW * D);
      // up-proj + GELU over gathered tokens of expert e
      gemm(h_bf, wb_w1, b1 + (size_t)(l * 4 + e) * HMAXW, nullptr, nullptr,
           hid_bf, S, counts + e, He, D, D, HMAXW, HMAXW, lists + e * S,
           nullptr, nullptr, 1);
      // down-proj, gate-scaled, scattered accumulate into residual stream
      gemm(hid_bf, wb_w2, w2b + (size_t)(l * 4 + e) * D, x_f32, x_f32, nullptr,
           S, counts + e, D, He, HMAXW, D, D, nullptr, lists + e * S,
           glists + e * S, 0);
    }
  }

  // ---- Final LayerNorm -> output (fp32) ----
  k_ln<<<S, 256, 0, stream>>>(x_f32, lnf_g, lnf_b, nullptr, (float*)d_out);
}
